// ResidualAttentionBlock_21337397526855
// MI455X (gfx1250) — matrix operand order repaired and checked by reading, not compiled, not run on hardware
//
#include <hip/hip_runtime.h>

typedef unsigned short u16;
typedef __attribute__((ext_vector_type(16))) __bf16 bf16x16;
typedef __attribute__((ext_vector_type(8)))  unsigned short u16x8;
typedef __attribute__((ext_vector_type(8)))  float f32x8;
typedef __attribute__((ext_vector_type(4)))  unsigned int u32x4;
typedef __attribute__((ext_vector_type(8)))  int i32x8;
typedef __attribute__((ext_vector_type(4)))  int i32x4;

#if defined(__has_builtin)
# if __has_builtin(__builtin_amdgcn_tensor_load_to_lds) && __has_builtin(__builtin_amdgcn_s_wait_tensorcnt)
#  define USE_TDM 1
# endif
#endif
#ifndef USE_TDM
# define USE_TDM 0
#endif

union Frag {
    bf16x16 b;
    u16x8   h[2];
};

__device__ __forceinline__ u16 f2bf(float f) {
    union { float f; unsigned u; } c; c.f = f;
    unsigned u = c.u;
    u = u + 0x7fffu + ((u >> 16) & 1u);     // round-to-nearest-even
    return (u16)(u >> 16);
}

__device__ __forceinline__ f32x8 fzero8() {
    f32x8 z = {0.f, 0.f, 0.f, 0.f, 0.f, 0.f, 0.f, 0.f};
    return z;
}

__device__ __forceinline__ f32x8 wmma_bf16(Frag a, Frag b, f32x8 c) {
    // D = A(16x32 bf16) * B(32x16 bf16) + C(16x16 f32)
    return __builtin_amdgcn_wmma_f32_16x16x32_bf16(
        /*neg_a=*/false, a.b, /*neg_b=*/false, b.b,
        /*c_mod=*/(short)0, c, /*reuse_a=*/false, /*reuse_b=*/false);
}

// TDM: load a (tile_rows x 32) bf16 tile from a row-major tensor (row stride =
// stride_elems) into LDS at lds_addr with a 16-byte pad after every 64-byte row
// (-> 80B LDS row stride == BKP=40 u16). Rows beyond rows_rem are zero-filled
// by the TDM's tensor-dim OOB clamp (handles the M tail for free).
__device__ __forceinline__ void tdm_load_tile(
    unsigned lds_addr, const u16* gptr, unsigned rows_rem,
    unsigned tile_rows, unsigned stride_elems)
{
#if USE_TDM
    unsigned long long ga = (unsigned long long)(size_t)gptr;
    u32x4 g0 = { 1u,                                   // count=1 (valid user D#)
                 lds_addr,                             // LDS byte address
                 (unsigned)ga,                         // global_addr[31:0]
                 ((unsigned)(ga >> 32) & 0x01ffffffu)  // global_addr[56:32]
                 | 0x80000000u };                      // type=2 ("image")
    if (rows_rem > 0xffffu) rows_rem = 0xffffu;
    i32x8 g1 = { (int)0x06D10000u,          // data_size=2B, pad_en, interval=16DW, amount=4DW
                 (int)(32u << 16),          // tensor_dim0[15:0] = 32
                 (int)(rows_rem << 16),     // tensor_dim0 hi=0 | tensor_dim1[15:0]
                 (int)(32u << 16),          // tensor_dim1 hi=0 | tile_dim0=32
                 (int)tile_rows,            // tile_dim1 | tile_dim2=0
                 (int)stride_elems,         // tensor_dim0_stride[31:0]
                 0, 0 };
    i32x4 gz4 = { 0, 0, 0, 0 };
    i32x8 gz8 = { 0, 0, 0, 0, 0, 0, 0, 0 };
    __builtin_amdgcn_tensor_load_to_lds(g0, g1, gz4, gz4, gz8, 0);
#else
    (void)lds_addr; (void)gptr; (void)rows_rem; (void)tile_rows; (void)stride_elems;
#endif
}

// ---------------------------------------------------------------- constants
#define DMODEL 768
#define NHEAD  12
#define HDIM   64
#define NQRY   100
#define NKV    257
#define NSEQ   357
#define NBS    32
#define NTOK   (NSEQ * NBS)   // 11424
#define MKV    (NKV * NBS)    // 8224
#define MQ     (NQRY * NBS)   // 3200

// ---------------------------------------------------------------- fp32 -> bf16
__global__ void cvt_bf16_kernel(const float* __restrict__ in, u16* __restrict__ out, int n) {
    int i = blockIdx.x * blockDim.x + threadIdx.x;
    int stride = gridDim.x * blockDim.x;
    for (; i < n; i += stride) out[i] = f2bf(in[i]);
}

// ---------------------------------------------------------------- LayerNorm (one wave per token)
__global__ __launch_bounds__(256) void ln_kernel(
    const float* __restrict__ x, const float* __restrict__ g,
    const float* __restrict__ bta, u16* __restrict__ out, int ntok)
{
    int wid = threadIdx.x >> 5, lane = threadIdx.x & 31;
    int tok = blockIdx.x * 8 + wid;
    if (tok >= ntok) return;
    const float* row = x + (size_t)tok * DMODEL;
    float v[24];
    float s = 0.f;
#pragma unroll
    for (int j = 0; j < 24; ++j) { v[j] = row[lane + 32 * j]; s += v[j]; }
#pragma unroll
    for (int o = 16; o >= 1; o >>= 1) s += __shfl_xor(s, o, 32);
    float mean = s * (1.f / DMODEL);
    float q = 0.f;
#pragma unroll
    for (int j = 0; j < 24; ++j) { float d = v[j] - mean; q += d * d; }
#pragma unroll
    for (int o = 16; o >= 1; o >>= 1) q += __shfl_xor(q, o, 32);
    float rstd = rsqrtf(q * (1.f / DMODEL) + 1e-5f);
    u16* orow = out + (size_t)tok * DMODEL;
#pragma unroll
    for (int j = 0; j < 24; ++j) {
        int c = lane + 32 * j;
        orow[c] = f2bf((v[j] - mean) * rstd * g[c] + bta[c]);
    }
}

// ---------------------------------------------------------------- WMMA GEMM
// out[M,N] = A[M,K](bf16) @ W[N,K]^T(bf16) + bias, fused epilogues:
// EPI 0: store bf16; EPI 1: QuickGELU then bf16; EPI 2: fp32 res[M,N] + v
// 128x64 block tile, 8 waves each computing 32x32 (4 WMMA / K-step).
// TDM path: one wave issues 2 tensor_load_to_lds per K-step into ping-pong
// LDS buffers, pipelined one stage ahead (s_wait_tensorcnt <= 2).
template <int EPI>
__global__ __launch_bounds__(256) void gemm_bf16_kernel(
    const u16* __restrict__ A, const u16* __restrict__ W,
    const float* __restrict__ bias, const float* __restrict__ res,
    void* __restrict__ outp, int M, int N, int K)
{
    constexpr int BM = 128, BN = 64, BK = 32, BKP = 40;
    __shared__ u16 As[2][BM * BKP];
    __shared__ u16 Ws[2][BN * BKP];

    int t = threadIdx.x, lane = t & 31;
    int wid = t >> 5;
    int half = lane >> 4, l16 = lane & 15;
    int waveM = wid & 3, waveN = wid >> 2;
    int m0 = blockIdx.x * BM, n0 = blockIdx.y * BN;

    f32x8 acc00 = fzero8(), acc01 = fzero8(), acc10 = fzero8(), acc11 = fzero8();
    int nk = K / BK;

#if USE_TDM
    if (t == 0) {
        tdm_load_tile((unsigned)(size_t)&As[0][0], A + (size_t)m0 * K, (unsigned)(M - m0), BM, (unsigned)K);
        tdm_load_tile((unsigned)(size_t)&Ws[0][0], W + (size_t)n0 * K, (unsigned)(N - n0), BN, (unsigned)K);
    }
#else
    int lrow = t >> 2;            // 0..63
    int lcol = (t & 3) * 8;       // 0,8,16,24
    bool a0ok = (m0 + lrow) < M;
    bool a1ok = (m0 + 64 + lrow) < M;
    const u16* Aptr0 = A + (size_t)(m0 + lrow) * K + lcol;
    const u16* Aptr1 = A + (size_t)(m0 + 64 + lrow) * K + lcol;
    const u16* Wptr  = W + (size_t)(n0 + lrow) * K + lcol;
#endif

    for (int ik = 0; ik < nk; ++ik) {
        int buf = ik & 1;
#if USE_TDM
        if (t == 0) {
            if (ik + 1 < nk) {
                int k1 = (ik + 1) * BK;
                tdm_load_tile((unsigned)(size_t)&As[buf ^ 1][0], A + (size_t)m0 * K + k1,
                              (unsigned)(M - m0), BM, (unsigned)K);
                tdm_load_tile((unsigned)(size_t)&Ws[buf ^ 1][0], W + (size_t)n0 * K + k1,
                              (unsigned)(N - n0), BN, (unsigned)K);
                __builtin_amdgcn_s_wait_tensorcnt(2);   // stage ik landed; ik+1 in flight
            } else {
                __builtin_amdgcn_s_wait_tensorcnt(0);
            }
        }
        __syncthreads();                                 // stage ik visible to all waves
#else
        {
            int k0 = ik * BK;
            u16x8 z = {0, 0, 0, 0, 0, 0, 0, 0};
            u16x8 av0 = z, av1 = z;
            if (a0ok) av0 = *(const u16x8*)(Aptr0 + k0);
            if (a1ok) av1 = *(const u16x8*)(Aptr1 + k0);
            u16x8 wv = *(const u16x8*)(Wptr + k0);
            *(u16x8*)(&As[buf][lrow * BKP + lcol]) = av0;
            *(u16x8*)(&As[buf][(64 + lrow) * BKP + lcol]) = av1;
            *(u16x8*)(&Ws[buf][lrow * BKP + lcol]) = wv;
        }
        __syncthreads();
#endif
        const u16* as = &As[buf][0];
        const u16* ws = &Ws[buf][0];
        // A fragments: row = l16, K elems {8h..8h+7, 16+8h..16+8h+7}
        Frag a0, a1, b0, b1;
        int am0 = (waveM * 32 + l16) * BKP;
        a0.h[0] = *(const u16x8*)(as + am0 + 8 * half);
        a0.h[1] = *(const u16x8*)(as + am0 + 16 + 8 * half);
        int am1 = (waveM * 32 + 16 + l16) * BKP;
        a1.h[0] = *(const u16x8*)(as + am1 + 8 * half);
        a1.h[1] = *(const u16x8*)(as + am1 + 16 + 8 * half);
        // B fragments: col = l16, K elems {8h..8h+7, 16+8h..16+8h+7}
        int bn0 = (waveN * 32 + l16) * BKP;
        b0.h[0] = *(const u16x8*)(ws + bn0 + 8 * half);
        b0.h[1] = *(const u16x8*)(ws + bn0 + 16 + 8 * half);
        int bn1 = (waveN * 32 + 16 + l16) * BKP;
        b1.h[0] = *(const u16x8*)(ws + bn1 + 8 * half);
        b1.h[1] = *(const u16x8*)(ws + bn1 + 16 + 8 * half);

        acc00 = wmma_bf16(a0, b0, acc00);
        acc01 = wmma_bf16(a0, b1, acc01);
        acc10 = wmma_bf16(a1, b0, acc10);
        acc11 = wmma_bf16(a1, b1, acc11);
        __syncthreads();                                 // readers done before buf reuse
    }

#pragma unroll
    for (int mi = 0; mi < 2; ++mi) {
#pragma unroll
        for (int nj = 0; nj < 2; ++nj) {
            f32x8 acc = mi ? (nj ? acc11 : acc10) : (nj ? acc01 : acc00);
            int col = n0 + waveN * 32 + nj * 16 + l16;
            float bv = bias[col];
#pragma unroll
            for (int i = 0; i < 8; ++i) {
                int row = m0 + waveM * 32 + mi * 16 + i + 8 * half;
                if (row >= M) continue;
                float v = acc[i] + bv;
                size_t idx = (size_t)row * N + col;
                if (EPI == 0) {
                    ((u16*)outp)[idx] = f2bf(v);
                } else if (EPI == 1) {
                    float gl = v / (1.f + __expf(-1.702f * v));   // QuickGELU
                    ((u16*)outp)[idx] = f2bf(gl);
                } else {
                    ((float*)outp)[idx] = res[idx] + v;
                }
            }
        }
    }
}

// ---------------------------------------------------------------- flash attention per (b,h)
// K/V^T staged in LDS once; each wave handles 16 queries with online softmax.
template <bool MASKED>
__global__ __launch_bounds__(256) void attn_kernel(
    const u16* __restrict__ qbuf, int qstride, int nq, int rowoff,
    const u16* __restrict__ qkv, const int* __restrict__ mask,
    u16* __restrict__ out)
{
    constexpr int KVP = 288;            // 257 padded to 9 tiles of 32
    constexpr int KSTR = 72;            // 144B row stride: 16B aligned, conflict-free
    constexpr int VSTR = 296;           // 592B row stride: 16B aligned, conflict-free
    constexpr int QKVS = 3 * DMODEL;
    __shared__ u16 Ks[KVP * KSTR];      // K [key][d]
    __shared__ u16 Vt[HDIM * VSTR];     // V^T [d][key]
    __shared__ u16 Pw[8][16 * 40];      // per-wave P bounce (C-layout -> A-layout)

    int b = blockIdx.x / NHEAD, h = blockIdx.x % NHEAD;
    int t = threadIdx.x, lane = t & 31, wid = t >> 5;
    int half = lane >> 4, l16 = lane & 15;

    // ---- stage K and V^T (zeros in the key-padding region)
    for (int c = t; c < KVP * 8; c += 256) {
        int row = c >> 3, col = (c & 7) * 8;
        u16x8 kv8 = {0, 0, 0, 0, 0, 0, 0, 0};
        u16x8 vv8 = {0, 0, 0, 0, 0, 0, 0, 0};
        if (row < NKV) {
            size_t base = ((size_t)(row * NBS + b)) * QKVS + h * HDIM + col;
            kv8 = *(const u16x8*)(qkv + base + DMODEL);       // K slice
            vv8 = *(const u16x8*)(qkv + base + 2 * DMODEL);   // V slice
        }
        *(u16x8*)(&Ks[row * KSTR + col]) = kv8;
#pragma unroll
        for (int j = 0; j < 8; ++j) Vt[(col + j) * VSTR + row] = vv8[j];
    }
    __syncthreads();

    const float scale = 0.125f;  // 1/sqrt(64), applied on scores (== scaling Q)

    for (int qb0 = 0; qb0 < nq; qb0 += 128) {
        int qb = qb0 + wid * 16;
        if (qb >= nq) continue;   // no barriers below: safe

        // ---- Q fragments for d = 0..31 and 32..63
        int qr = qb + l16; if (qr > nq - 1) qr = nq - 1;
        size_t qbase = ((size_t)(qr * NBS + b)) * qstride + h * HDIM;
        Frag a0, a1;
        a0.h[0] = *(const u16x8*)(qbuf + qbase + 8 * half);
        a0.h[1] = *(const u16x8*)(qbuf + qbase + 16 + 8 * half);
        a1.h[0] = *(const u16x8*)(qbuf + qbase + 32 + 8 * half);
        a1.h[1] = *(const u16x8*)(qbuf + qbase + 48 + 8 * half);

        f32x8 o0 = fzero8(), o1 = fzero8(), o2 = fzero8(), o3 = fzero8();
        float runmax[8], runsum[8];
#pragma unroll
        for (int i = 0; i < 8; ++i) { runmax[i] = -1e30f; runsum[i] = 0.f; }

        for (int m0 = 0; m0 < KVP; m0 += 32) {
            // ---- scores S[16q x 32m] via 4 WMMA
            f32x8 s0 = fzero8(), s1 = fzero8();
            {
                Frag bk;
                int km = m0 + l16;
                bk.h[0] = *(const u16x8*)(&Ks[km * KSTR + 8 * half]);
                bk.h[1] = *(const u16x8*)(&Ks[km * KSTR + 16 + 8 * half]);
                s0 = wmma_bf16(a0, bk, s0);
                bk.h[0] = *(const u16x8*)(&Ks[km * KSTR + 32 + 8 * half]);
                bk.h[1] = *(const u16x8*)(&Ks[km * KSTR + 48 + 8 * half]);
                s0 = wmma_bf16(a1, bk, s0);
                km = m0 + 16 + l16;
                bk.h[0] = *(const u16x8*)(&Ks[km * KSTR + 8 * half]);
                bk.h[1] = *(const u16x8*)(&Ks[km * KSTR + 16 + 8 * half]);
                s1 = wmma_bf16(a0, bk, s1);
                bk.h[0] = *(const u16x8*)(&Ks[km * KSTR + 32 + 8 * half]);
                bk.h[1] = *(const u16x8*)(&Ks[km * KSTR + 48 + 8 * half]);
                s1 = wmma_bf16(a1, bk, s1);
            }

            int ma = m0 + l16, mb = m0 + 16 + l16;
#pragma unroll
            for (int i = 0; i < 8; ++i) {
                int qi = qb + i + 8 * half;
                float x0 = s0[i] * scale, x1 = s1[i] * scale;
                bool v0 = (ma < NKV), v1 = (mb < NKV);
                if (MASKED) {
                    int qc = qi < nq ? qi : nq - 1;
                    size_t mbase = ((size_t)b * nq + qc) * NKV;
                    if (v0 && mask[mbase + ma]) v0 = false;
                    if (v1 && mask[mbase + mb]) v1 = false;
                }
                if (!v0) x0 = -1e30f;
                if (!v1) x1 = -1e30f;
                // row max over 16 lanes (same half-wave)
                float tm = fmaxf(x0, x1);
                tm = fmaxf(tm, __shfl_xor(tm, 1, 32));
                tm = fmaxf(tm, __shfl_xor(tm, 2, 32));
                tm = fmaxf(tm, __shfl_xor(tm, 4, 32));
                tm = fmaxf(tm, __shfl_xor(tm, 8, 32));
                float nm = fmaxf(runmax[i], tm);
                float sc = __expf(runmax[i] - nm);
                runmax[i] = nm;
                float p0 = __expf(x0 - nm), p1 = __expf(x1 - nm);
                float ts = p0 + p1;
                ts += __shfl_xor(ts, 1, 32);
                ts += __shfl_xor(ts, 2, 32);
                ts += __shfl_xor(ts, 4, 32);
                ts += __shfl_xor(ts, 8, 32);
                runsum[i] = runsum[i] * sc + ts;
                o0[i] *= sc; o1[i] *= sc; o2[i] *= sc; o3[i] *= sc;
                int prow = i + 8 * half;
                Pw[wid][prow * 40 + l16]      = f2bf(p0);
                Pw[wid][prow * 40 + 16 + l16] = f2bf(p1);
            }

            // ---- P @ V via 4 WMMA (P bounced through LDS into A-layout; same-wave
            // LDS ops are in-order so no barrier needed)
            Frag pa;
            pa.h[0] = *(const u16x8*)(&Pw[wid][l16 * 40 + 8 * half]);
            pa.h[1] = *(const u16x8*)(&Pw[wid][l16 * 40 + 16 + 8 * half]);
            Frag bv;
#pragma unroll
            for (int dt = 0; dt < 4; ++dt) {
                const u16* vr = &Vt[(dt * 16 + l16) * VSTR + m0 + 8 * half];
                bv.h[0] = *(const u16x8*)(vr);
                bv.h[1] = *(const u16x8*)(vr + 16);
                if (dt == 0) o0 = wmma_bf16(pa, bv, o0);
                else if (dt == 1) o1 = wmma_bf16(pa, bv, o1);
                else if (dt == 2) o2 = wmma_bf16(pa, bv, o2);
                else o3 = wmma_bf16(pa, bv, o3);
            }
        }

        // ---- normalize and store bf16
#pragma unroll
        for (int i = 0; i < 8; ++i) {
            int qi = qb + i + 8 * half;
            if (qi >= nq) continue;
            float inv = 1.f / runsum[i];
            size_t tokbase = ((size_t)((rowoff + qi) * NBS + b)) * DMODEL + h * HDIM;
            out[tokbase + 0 * 16 + l16] = f2bf(o0[i] * inv);
            out[tokbase + 1 * 16 + l16] = f2bf(o1[i] * inv);
            out[tokbase + 2 * 16 + l16] = f2bf(o2[i] * inv);
            out[tokbase + 3 * 16 + l16] = f2bf(o3[i] * inv);
        }
    }
}

// ---------------------------------------------------------------- host
extern "C" void kernel_launch(void* const* d_in, const int* in_sizes, int n_in,
                              void* d_out, int out_size, void* d_ws, size_t ws_size,
                              hipStream_t stream) {
    (void)in_sizes; (void)n_in; (void)out_size; (void)ws_size;

    const float* y         = (const float*)d_in[0];
    const int*   attn_mask = (const int*)d_in[1];
    const float* in_proj_w = (const float*)d_in[2];
    const float* in_proj_b = (const float*)d_in[3];
    const float* new_q_w   = (const float*)d_in[4];
    const float* new_q_b   = (const float*)d_in[5];
    const float* out_w     = (const float*)d_in[6];
    const float* out_b     = (const float*)d_in[7];
    const float* ln1_g     = (const float*)d_in[8];
    const float* ln1_b     = (const float*)d_in[9];
    const float* ln2_g     = (const float*)d_in[10];
    const float* ln2_b     = (const float*)d_in[11];
    const float* fc_w      = (const float*)d_in[12];
    const float* fc_b      = (const float*)d_in[13];
    const float* proj_w    = (const float*)d_in[14];
    const float* proj_b    = (const float*)d_in[15];

    // workspace layout (bf16 regions; region C is reused for qkv/newq/attn then h)
    u16* w_inproj = (u16*)d_ws;
    u16* w_newq   = w_inproj + (size_t)3 * DMODEL * DMODEL;
    u16* w_out    = w_newq   + (size_t)DMODEL * DMODEL;
    u16* w_fc     = w_out    + (size_t)DMODEL * DMODEL;
    u16* w_proj   = w_fc     + (size_t)4 * DMODEL * DMODEL;
    u16* xbf      = w_proj   + (size_t)4 * DMODEL * DMODEL;          // region B (x, then LN2 out)
    u16* regionC  = xbf      + (size_t)NTOK * DMODEL;
    u16* qkv      = regionC;
    u16* nqb      = qkv + (size_t)MKV * 3 * DMODEL;
    u16* attn     = nqb + (size_t)MQ * DMODEL;
    u16* hbuf     = regionC;                                          // alias (later phase)
    float* y2     = (float*)(regionC + (size_t)NTOK * 4 * DMODEL);

    // 1) weights -> bf16
    cvt_bf16_kernel<<<1024, 256, 0, stream>>>(in_proj_w, w_inproj, 3 * DMODEL * DMODEL);
    cvt_bf16_kernel<<<1024, 256, 0, stream>>>(new_q_w,   w_newq,   DMODEL * DMODEL);
    cvt_bf16_kernel<<<1024, 256, 0, stream>>>(out_w,     w_out,    DMODEL * DMODEL);
    cvt_bf16_kernel<<<1024, 256, 0, stream>>>(fc_w,      w_fc,     4 * DMODEL * DMODEL);
    cvt_bf16_kernel<<<1024, 256, 0, stream>>>(proj_w,    w_proj,   4 * DMODEL * DMODEL);

    // 2) LN1: y -> x(bf16)  [rows 0..MQ are query tokens, MQ.. are kv tokens]
    ln_kernel<<<(NTOK + 7) / 8, 256, 0, stream>>>(y, ln1_g, ln1_b, xbf, NTOK);

    // 3) qkv = x_kv @ in_proj^T + b  (bf16 out)
    gemm_bf16_kernel<0><<<dim3((MKV + 127) / 128, (3 * DMODEL) / 64), 256, 0, stream>>>(
        xbf + (size_t)MQ * DMODEL, w_inproj, in_proj_b, nullptr, qkv, MKV, 3 * DMODEL, DMODEL);

    // 4) new_q = x_q @ new_q^T + b  (bf16 out)
    gemm_bf16_kernel<0><<<dim3((MQ + 127) / 128, DMODEL / 64), 256, 0, stream>>>(
        xbf, w_newq, new_q_b, nullptr, nqb, MQ, DMODEL, DMODEL);

    // 5) attention: masked queries (new_q) + clip queries (q slice of qkv)
    attn_kernel<true><<<NBS * NHEAD, 256, 0, stream>>>(
        nqb, DMODEL, NQRY, 0, qkv, attn_mask, attn);
    attn_kernel<false><<<NBS * NHEAD, 256, 0, stream>>>(
        qkv, 3 * DMODEL, NKV, NQRY, qkv, attn_mask, attn);

    // 6) y2 = y + attn @ out_w^T + b  (fp32, fused residual)
    gemm_bf16_kernel<2><<<dim3((NTOK + 127) / 128, DMODEL / 64), 256, 0, stream>>>(
        attn, w_out, out_b, y, y2, NTOK, DMODEL, DMODEL);

    // 7) LN2: y2 -> xbf (bf16, region B reused)
    ln_kernel<<<(NTOK + 7) / 8, 256, 0, stream>>>(y2, ln2_g, ln2_b, xbf, NTOK);

    // 8) h = QuickGELU(xbf @ fc^T + b)  (bf16, region C reused)
    gemm_bf16_kernel<1><<<dim3((NTOK + 127) / 128, (4 * DMODEL) / 64), 256, 0, stream>>>(
        xbf, w_fc, fc_b, nullptr, hbuf, NTOK, 4 * DMODEL, DMODEL);

    // 9) out = y2 + h @ proj^T + b  (fp32 -> d_out)
    gemm_bf16_kernel<2><<<dim3((NTOK + 127) / 128, DMODEL / 64), 256, 0, stream>>>(
        hbuf, w_proj, proj_b, y2, d_out, NTOK, DMODEL, 4 * DMODEL);
}
